// Attention_8985071583691
// MI455X (gfx1250) — matrix operand order repaired and checked by reading, not compiled, not run on hardware
//
#include <hip/hip_runtime.h>

typedef __bf16 bf16;
typedef __bf16 v16bf __attribute__((ext_vector_type(16)));
typedef __bf16 v8bf  __attribute__((ext_vector_type(8)));
typedef __bf16 v2bf  __attribute__((ext_vector_type(2)));
typedef float  v8f   __attribute__((ext_vector_type(8)));
typedef int    v4i   __attribute__((ext_vector_type(4)));

#define B_SZ 4
#define L_SZ 2048
#define S_SZ 2048
#define H_SZ 16
#define E_SZ 64
#define D_SZ 1024
#define BM   128   // query rows per workgroup (16 per wave x 8 waves)
#define BN   64    // key rows per block iteration (4 x 16-row score tiles)

__device__ __forceinline__ v16bf cat8(v8bf a, v8bf b) {
  return __builtin_shufflevector(a, b, 0,1,2,3,4,5,6,7,8,9,10,11,12,13,14,15);
}

// swap the 16-byte v8bf across the lane pair (lane ^ 16) with 4 dword shuffles
__device__ __forceinline__ v8bf swap16(v8bf x) {
  v4i w = __builtin_bit_cast(v4i, x), r;
  #pragma unroll
  for (int i = 0; i < 4; ++i) r[i] = __shfl_xor(w[i], 16, 32);
  return __builtin_bit_cast(v8bf, r);
}

__device__ __forceinline__ v8bf cvt8(float4 a, float4 b) {
  v8bf o;
  o[0]=(bf16)a.x; o[1]=(bf16)a.y; o[2]=(bf16)a.z; o[3]=(bf16)a.w;
  o[4]=(bf16)b.x; o[5]=(bf16)b.y; o[6]=(bf16)b.z; o[7]=(bf16)b.w;
  return o;
}

// stage two s-rows x 8 e: K row-major (b128 stores), V transposed (b32 pair stores)
__device__ __forceinline__ void stage_tile(bf16* kdst, bf16* vdst, int sr2, int scol,
                                           const float4* kf, const float4* vf)
{
  *(v8bf*)&kdst[ sr2      * E_SZ + scol] = cvt8(kf[0], kf[1]);
  *(v8bf*)&kdst[(sr2 + 1) * E_SZ + scol] = cvt8(kf[2], kf[3]);
  const float* v0 = (const float*)&vf[0];   // row sr2,   e scol..scol+7
  const float* v1 = (const float*)&vf[2];   // row sr2+1, e scol..scol+7
  #pragma unroll
  for (int i = 0; i < 8; ++i) {
    v2bf pr; pr[0] = (bf16)v0[i]; pr[1] = (bf16)v1[i];
    *(v2bf*)&vdst[(scol + i) * BN + sr2] = pr;          // ds_store_b32
  }
}

__global__ __launch_bounds__(256)
void fa_fwd_cdna5(const float* __restrict__ Q, const float* __restrict__ K,
                  const float* __restrict__ V, float* __restrict__ O)
{
  // double-buffered tiles: K [s][e] row-major, V^T [e][s]; 2*(8+8) KB = 32 KB
  __shared__ __align__(16) bf16 lds_k [2][BN * E_SZ];
  __shared__ __align__(16) bf16 lds_vt[2][E_SZ * BN];

  const int tid  = threadIdx.x;
  // assert wave-uniformity so causal guards compile to SGPR branches
  // (EXEC provably all-ones at every WMMA, no saveexec chains)
  const int wave = __builtin_amdgcn_readfirstlane(tid >> 5);
  const int lane = tid & 31;
  const int lh   = lane & 15;
  const bool hi  = lane >= 16;

  const int q0   = blockIdx.x * BM;
  const int b    = blockIdx.y >> 4;          // H_SZ == 16
  const int h    = blockIdx.y & 15;
  const int qw   = q0 + wave * 16;           // scalar (uniform)
  const int qcol = qw + lh;                  // this lane's query column

  const size_t headQ = (size_t)b * L_SZ * D_SZ + (size_t)h * E_SZ;
  const size_t headK = (size_t)b * S_SZ * D_SZ + (size_t)h * E_SZ;

  constexpr float CF = 0.125f * 1.44269504088896340736f; // 1/sqrt(E) * log2(e)

  // ---- Q as WMMA B operand (32x16): lane = column q; lo lanes K(e) 0..7,16..23,
  //      hi lanes 8..15,24..31 per 32-e chunk. Softmax scale folded in here. ----
  v16bf bq[2];
  {
    const float* qp = Q + headQ + (size_t)qcol * D_SZ;
    #pragma unroll
    for (int c = 0; c < 2; ++c) {
      const float* p = qp + c * 32 + (hi ? 8 : 0);
      #pragma unroll
      for (int i = 0; i < 16; ++i) bq[c][i] = (bf16)(p[i + (i & 8)] * CF);
    }
  }

  // O^T accumulator tiles (e x q, C-layout: lane = q, VGPR r -> e = 16c + r + 8*hi)
  v8f acc[4];
  #pragma unroll
  for (int c = 0; c < 4; ++c) acc[c] = (v8f){};
  float mrun = -1e30f, lrun = 0.0f;          // scalar per-lane softmax state

  const int nblk   = q0 / BN + BM / BN;      // workgroup causal frontier (scalar)
  const int nblk_w = (qw + 15) / BN + 1;     // this wave's frontier (scalar)

  const int sr2  = (tid >> 3) << 1;          // staging: even key row 0..62
  const int scol = (tid & 7) << 3;           // staging: e 0,8,..,56
  const float* kbase = K + headK + (size_t)sr2 * D_SZ + scol;
  const float* vbase = V + headK + (size_t)sr2 * D_SZ + scol;

  // ---- stage block 0 into buffer 0 ----
  {
    float4 kf[4], vf[4];
    kf[0]=((const float4*)kbase)[0]; kf[1]=((const float4*)kbase)[1];
    kf[2]=((const float4*)(kbase+D_SZ))[0]; kf[3]=((const float4*)(kbase+D_SZ))[1];
    vf[0]=((const float4*)vbase)[0]; vf[1]=((const float4*)vbase)[1];
    vf[2]=((const float4*)(vbase+D_SZ))[0]; vf[3]=((const float4*)(vbase+D_SZ))[1];
    stage_tile(lds_k[0], lds_vt[0], sr2, scol, kf, vf);
  }
  __syncthreads();

  for (int blk = 0; blk < nblk; ++blk) {
    const int buf = blk & 1;
    const int s0  = blk * BN;                // scalar
    const bool have_next = (blk + 1 < nblk);

    // ---- issue next tile's global loads before compute (double buffering) ----
    float4 kf[4], vf[4];
    if (have_next) {
      const float* kp = kbase + (size_t)(blk + 1) * BN * D_SZ;
      const float* vp = vbase + (size_t)(blk + 1) * BN * D_SZ;
      kf[0]=((const float4*)kp)[0]; kf[1]=((const float4*)kp)[1];
      kf[2]=((const float4*)(kp+D_SZ))[0]; kf[3]=((const float4*)(kp+D_SZ))[1];
      vf[0]=((const float4*)vp)[0]; vf[1]=((const float4*)vp)[1];
      vf[2]=((const float4*)(vp+D_SZ))[0]; vf[3]=((const float4*)(vp+D_SZ))[1];
      if (blk + 2 < nblk) {
        __builtin_prefetch(kp + (size_t)BN * D_SZ, 0, 1);   // global_prefetch_b8
        __builtin_prefetch(vp + (size_t)BN * D_SZ, 0, 1);
      }
    }

    if (blk < nblk_w) {                      // scalar causal block skip
      // ---- S^T = K_tile * Q^T : four 16-row s-tiles, per-tile scalar guards ----
      const bf16* kb = lds_k[buf];
      v8f sc[4];
      #pragma unroll
      for (int n = 0; n < 4; ++n) {
        #pragma unroll
        for (int r = 0; r < 8; ++r) sc[n][r] = -1e30f;   // default: fully masked
        if (s0 + 16 * n <= qw + 15) {        // scalar: tile has valid elements
          v8f cz = (v8f){};
          #pragma unroll
          for (int c = 0; c < 2; ++c) {
            const bf16* p = &kb[(n * 16 + lh) * E_SZ + c * 32 + (hi ? 8 : 0)];
            v16bf ak = cat8(*(const v8bf*)p, *(const v8bf*)(p + 16));
            cz = __builtin_amdgcn_wmma_f32_16x16x32_bf16(
                   false, ak, false, bq[c], (short)0, cz, false, false);
          }
          if (s0 + 16 * n + 15 > qw) {       // scalar: tile straddles the diagonal
            #pragma unroll
            for (int r = 0; r < 8; ++r) {
              const int si = s0 + 16 * n + (hi ? 8 : 0) + r;
              cz[r] = (si <= qcol) ? cz[r] : -1e30f;
            }
          }
          sc[n] = cz;
        }
      }

      // ---- online softmax over s: in-lane tree + one partner-lane exchange ----
      float t01 = fmaxf(sc[0][0], sc[1][0]);
      float t23 = fmaxf(sc[2][0], sc[3][0]);
      #pragma unroll
      for (int r = 1; r < 8; ++r) {
        t01 = fmaxf(t01, fmaxf(sc[0][r], sc[1][r]));
        t23 = fmaxf(t23, fmaxf(sc[2][r], sc[3][r]));
      }
      float t = fmaxf(t01, t23);
      t = fmaxf(t, __shfl_xor(t, 16, 32));
      const float mnew  = fmaxf(mrun, t);
      const float alpha = exp2f(mrun - mnew);          // scale already folded in
      mrun = mnew;

      float pv[4][8], rs = 0.0f;
      #pragma unroll
      for (int n = 0; n < 4; ++n)
        #pragma unroll
        for (int r = 0; r < 8; ++r) {
          pv[n][r] = exp2f(sc[n][r] - mnew);
          rs += pv[n][r];
        }
      rs += __shfl_xor(rs, 16, 32);
      lrun = lrun * alpha + rs;
      #pragma unroll
      for (int c = 0; c < 4; ++c) acc[c] = acc[c] * alpha;   // v_pk_mul_f32-able

      // ---- O^T += V^T * P^T : two 32-key halves; P^T built via packed swap ----
      const bf16* vtb = lds_vt[buf];
      #pragma unroll
      for (int ss = 0; ss < 2; ++ss) {
        if (s0 + 32 * ss <= qw + 15) {       // scalar: skip fully-masked half
          v8bf pa, pb;
          #pragma unroll
          for (int r = 0; r < 8; ++r) {
            pa[r] = (bf16)pv[2 * ss][r];     // own s-subtile (lo half rows)
            pb[r] = (bf16)pv[2 * ss + 1][r]; // own s-subtile (hi half rows)
          }
          const v8bf pax = swap16(pa), pbx = swap16(pb);
          // lo lane: K 0..15 = own pa | partner pa ; hi: K 16..31 = partner pb | own pb
          const v16bf bp = hi ? cat8(pbx, pb) : cat8(pa, pax);
          #pragma unroll
          for (int c = 0; c < 4; ++c) {
            const bf16* p = &vtb[(c * 16 + lh) * BN + 32 * ss + (hi ? 16 : 0)];
            v16bf av = cat8(*(const v8bf*)p, *(const v8bf*)(p + 8));
            acc[c] = __builtin_amdgcn_wmma_f32_16x16x32_bf16(
                       false, av, false, bp, (short)0, acc[c], false, false);
          }
        }
      }
    }

    // ---- convert + store next tile into the other buffer, then one barrier ----
    if (have_next)
      stage_tile(lds_k[buf ^ 1], lds_vt[buf ^ 1], sr2, scol, kf, vf);
    __syncthreads();
  }

  // ---- epilogue: lane pair (lo e0..7 | hi e8..15) covers full 64B lines ----
  const float inv = 1.0f / lrun;
  float* op = O + headQ + (size_t)qcol * D_SZ + (hi ? 8 : 0);
  #pragma unroll
  for (int c = 0; c < 4; ++c) {
    float4 s0v, s1v;
    s0v.x = acc[c][0] * inv; s0v.y = acc[c][1] * inv;
    s0v.z = acc[c][2] * inv; s0v.w = acc[c][3] * inv;
    s1v.x = acc[c][4] * inv; s1v.y = acc[c][5] * inv;
    s1v.z = acc[c][6] * inv; s1v.w = acc[c][7] * inv;
    ((float4*)(op + c * 16))[0]     = s0v;   // global_store_b128
    ((float4*)(op + c * 16 + 4))[0] = s1v;   // global_store_b128
  }
}

extern "C" void kernel_launch(void* const* d_in, const int* in_sizes, int n_in,
                              void* d_out, int out_size, void* d_ws, size_t ws_size,
                              hipStream_t stream) {
  const float* q = (const float*)d_in[0];
  const float* k = (const float*)d_in[1];
  const float* v = (const float*)d_in[2];
  // d_in[3] is the attn_mask (pure causal) — reproduced analytically in-kernel.
  float* out = (float*)d_out;
  dim3 grid(L_SZ / BM, B_SZ * H_SZ);   // (16, 64)
  fa_fwd_cdna5<<<grid, 256, 0, stream>>>(q, k, v, out);
}
